// TransformerStage_91336774517018
// MI455X (gfx1250) — matrix operand order repaired and checked by reading, not compiled, not run on hardware
//
#include <hip/hip_runtime.h>

// ---------------- problem constants ----------------
#define LAYERS 6
#define BATCH  8
#define NHEAD  16
#define FDIM   1024
#define FH     64
#define NTOK   577
#define HIDDEN 4096
#define MTOT   (BATCH * NTOK)        // 4616 tokens total
#define NPAD   608                   // padded key length (19 chunks of 32)
#define NRDIST 2212                  // (2*24-1)^2 + 3
#define QTILES ((NTOK + 15) / 16)    // 37 query tiles of 16

typedef __attribute__((ext_vector_type(16))) __bf16 v16bf;
typedef __attribute__((ext_vector_type(8)))  float  v8f;
typedef __attribute__((ext_vector_type(4)))  unsigned int u32x4;
typedef __attribute__((ext_vector_type(8)))  int    i32x8;
typedef __attribute__((ext_vector_type(4)))  int    i32x4;

union Frag {            // one WMMA bf16 operand: 16 bf16 / lane = 2x uint4
    v16bf v;
    uint4 u[2];
};

static __device__ __forceinline__ unsigned short f2bf(float f) {
    unsigned int u = __float_as_uint(f);
    u += 0x7fffu + ((u >> 16) & 1u);     // round to nearest even
    return (unsigned short)(u >> 16);
}

// ---- Tensor Data Mover: 2-D tile load (global -> LDS), padded rows ----
// D# per CDNA5 ISA ch.8: data_size=2B, pad_enable, pad_interval=16 DW (64B row),
// pad_amount=4 DW (16B) -> LDS row stride 80B. OOB rows/cols read as zero.
static __device__ __forceinline__ void tdm_load_2d(
        unsigned lds_addr, unsigned long long gaddr,
        unsigned tdim0, unsigned tdim1,      // remaining tensor extent (elems)
        unsigned tile0, unsigned tile1,      // tile dims (elems)
        unsigned stride0) {                  // row stride (elems)
    u32x4 g0;
    g0.x = 1u;                                                // count=1
    g0.y = lds_addr;                                          // LDS byte addr
    g0.z = (unsigned)gaddr;                                   // global[31:0]
    g0.w = (unsigned)((gaddr >> 32) & 0x1FFFFFFull) | 0x80000000u; // +type=2
    i32x8 g1;
    g1[0] = (int)((1u << 16) | (1u << 20) | (3u << 22) | (3u << 25));
    g1[1] = (int)((tdim0 & 0xFFFFu) << 16);
    g1[2] = (int)((tdim0 >> 16) | ((tdim1 & 0xFFFFu) << 16));
    g1[3] = (int)((tdim1 >> 16) | (tile0 << 16));
    g1[4] = (int)(tile1 & 0xFFFFu);
    g1[5] = (int)stride0;
    g1[6] = 0;
    g1[7] = 0;
    i32x4 z4 = (i32x4)0;
#if __clang_major__ >= 23
    i32x8 z8 = (i32x8)0;
    __builtin_amdgcn_tensor_load_to_lds(g0, g1, z4, z4, z8, 0);
#else
    __builtin_amdgcn_tensor_load_to_lds(g0, g1, z4, z4, 0);
#endif
}

// BEiT relative position bias lookup (REL_IDX derived analytically)
static __device__ __forceinline__ float relbias(const float* __restrict__ tab,
                                                int h, int n, int m) {
    int idx;
    if (m == 0)      idx = (n == 0) ? (NRDIST - 1) : (NRDIST - 2);
    else if (n == 0) idx = NRDIST - 3;
    else {
        int pn = n - 1, pm = m - 1;
        int dy = (pn / 24) - (pm / 24) + 23;
        int dx = (pn % 24) - (pm % 24) + 23;
        idx = dy * 47 + dx;
    }
    return tab[idx * NHEAD + h];
}

// ---------------- f32 -> bf16 weight conversion (per-GEMM prepass) ----------------
__global__ __launch_bounds__(256) void cvt_kernel(const float4* __restrict__ src,
                                                  uint2* __restrict__ dst, int n4) {
    int i = blockIdx.x * 256 + threadIdx.x;
    if (i >= n4) return;
    float4 f = src[i];
    union { unsigned short h[4]; uint2 u; } t;
    t.h[0] = f2bf(f.x); t.h[1] = f2bf(f.y);
    t.h[2] = f2bf(f.z); t.h[3] = f2bf(f.w);
    dst[i] = t.u;
}

// ---------------- LayerNorm -> bf16 activations ----------------
__global__ __launch_bounds__(256) void ln_kernel(const float* __restrict__ x,
                                                 const float* __restrict__ g,
                                                 const float* __restrict__ bta,
                                                 unsigned short* __restrict__ o) {
    __shared__ float red[256];
    int row = blockIdx.x;
    int tid = threadIdx.x;
    const float* xr = x + (size_t)row * FDIM;
    float v[4];
    float s = 0.f;
#pragma unroll
    for (int i = 0; i < 4; i++) { v[i] = xr[tid + i * 256]; s += v[i]; }
    red[tid] = s;
    __syncthreads();
    for (int d = 128; d > 0; d >>= 1) {
        if (tid < d) red[tid] += red[tid + d];
        __syncthreads();
    }
    float mean = red[0] * (1.0f / FDIM);
    __syncthreads();
    float s2 = 0.f;
#pragma unroll
    for (int i = 0; i < 4; i++) { float t = v[i] - mean; s2 += t * t; }
    red[tid] = s2;
    __syncthreads();
    for (int d = 128; d > 0; d >>= 1) {
        if (tid < d) red[tid] += red[tid + d];
        __syncthreads();
    }
    float rstd = rsqrtf(red[0] * (1.0f / FDIM) + 1e-6f);
    unsigned short* orow = o + (size_t)row * FDIM;
#pragma unroll
    for (int i = 0; i < 4; i++) {
        int c = tid + i * 256;
        orow[c] = f2bf((v[i] - mean) * rstd * g[c] + bta[c]);
    }
}

// ---------------- WMMA GEMM: C[M,NC] = Abf16[M,K] @ Wbf16[NC,K]^T ----------------
// TDM double-buffered LDS staging. Wave 0 drives the DMA; TENSORcnt + barrier
// publish tile completion. LDS row stride 80B via TDM hardware padding.
// mode 0: QKV split epilogue  mode 1: x += scale*(acc+bias)  mode 2: gelu->bf16
__global__ __launch_bounds__(256) void gemm_kernel(
        const unsigned short* __restrict__ A, const unsigned short* __restrict__ W,
        int K, int NC, int mode,
        const float* __restrict__ bias, const float* __restrict__ scale,
        float* __restrict__ xres, unsigned short* __restrict__ obf,
        const float* __restrict__ qb, const float* __restrict__ vb,
        unsigned short* __restrict__ qo, unsigned short* __restrict__ ko,
        unsigned short* __restrict__ vto) {
    __shared__ __align__(16) unsigned short As[2][128][40];
    __shared__ __align__(16) unsigned short Bs[2][64][40];
    int tid = threadIdx.x;
    int lane = tid & 31, w = tid >> 5;
    int wm = w >> 1, wn = w & 1;
    int hlf = lane >> 4, l16 = lane & 15;
    int m0 = blockIdx.x * 128;
    int n0 = blockIdx.y * 64;
    v8f acc[2][2] = {};

    int nk = K >> 5;
    unsigned long long abase =
        (unsigned long long)(uintptr_t)(A + (size_t)m0 * K);
    unsigned long long bbase =
        (unsigned long long)(uintptr_t)(W + (size_t)n0 * K);
    unsigned rowsA = (unsigned)(MTOT - m0);     // OOB rows -> zero fill by TDM
    bool issuer = (w == 0);

    if (issuer) {
        tdm_load_2d((unsigned)(uintptr_t)&As[0][0][0], abase,
                    (unsigned)K, rowsA, 32u, 128u, (unsigned)K);
        tdm_load_2d((unsigned)(uintptr_t)&Bs[0][0][0], bbase,
                    (unsigned)K, 64u, 32u, 64u, (unsigned)K);
    }

    for (int kt = 0; kt < nk; kt++) {
        int cur = kt & 1;
        if (issuer) {
            if (kt + 1 < nk) {
                unsigned long long koff = (unsigned long long)(kt + 1) * 64ull;
                unsigned krem = (unsigned)(K - (kt + 1) * 32);
                tdm_load_2d((unsigned)(uintptr_t)&As[cur ^ 1][0][0],
                            abase + koff, krem, rowsA, 32u, 128u, (unsigned)K);
                tdm_load_2d((unsigned)(uintptr_t)&Bs[cur ^ 1][0][0],
                            bbase + koff, krem, 64u, 32u, 64u, (unsigned)K);
                __builtin_amdgcn_s_wait_tensorcnt(2);   // current pair done
            } else {
                __builtin_amdgcn_s_wait_tensorcnt(0);
            }
        }
        __syncthreads();    // publish buf[cur]; prior iter's readers are done

        Frag a[2], b[2];
#pragma unroll
        for (int i = 0; i < 2; i++) {            // A-frag lane layout
            int row = wm * 32 + i * 16 + l16;
            int c0 = hlf * 8;
            a[i].u[0] = *(const uint4*)(&As[cur][row][c0]);
            a[i].u[1] = *(const uint4*)(&As[cur][row][c0 + 16]);
        }
#pragma unroll
        for (int j = 0; j < 2; j++) {            // B-frag lane layout
            int col = wn * 32 + j * 16 + l16;
            int c0 = hlf * 8;
            b[j].u[0] = *(const uint4*)(&Bs[cur][col][c0]);
            b[j].u[1] = *(const uint4*)(&Bs[cur][col][c0 + 16]);
        }
#pragma unroll
        for (int i = 0; i < 2; i++)
#pragma unroll
            for (int j = 0; j < 2; j++)
                acc[i][j] = __builtin_amdgcn_wmma_f32_16x16x32_bf16(
                    false, a[i].v, false, b[j].v, (short)0, acc[i][j], false, false);
        __syncthreads();    // done reading buf[cur] before DMA reuses it
    }

    // ---- epilogue ----
#pragma unroll
    for (int i = 0; i < 2; i++)
#pragma unroll
        for (int j = 0; j < 2; j++)
#pragma unroll
            for (int r = 0; r < 8; r++) {
                int row = m0 + wm * 32 + i * 16 + r + hlf * 8;
                int col = n0 + wn * 32 + j * 16 + l16;
                if (row >= MTOT) continue;
                float v = acc[i][j][r];
                if (mode == 0) {
                    int three = col >> 10, rem = col & 1023;
                    int h = rem >> 6, f = rem & 63;
                    int bb = row / NTOK, n = row % NTOK;
                    size_t bh = (size_t)(bb * NHEAD + h);
                    if (three == 0) {
                        float q = (v + qb[h * 64 + f]) * 0.125f;   // qk_scale
                        qo[(bh * NPAD + n) * 64 + f] = f2bf(q);
                    } else if (three == 1) {
                        ko[(bh * NPAD + n) * 64 + f] = f2bf(v);
                    } else {
                        float vv = v + vb[h * 64 + f];
                        vto[(bh * 64 + f) * NPAD + n] = f2bf(vv);  // V transposed
                    }
                } else if (mode == 1) {
                    float o = v + bias[col];
                    xres[(size_t)row * NC + col] += scale[col] * o;
                } else {
                    float o = v + bias[col];
                    o = 0.5f * o * (1.0f + erff(o * 0.70710678118654752f));
                    obf[(size_t)row * NC + col] = f2bf(o);
                }
            }
}

// ---------------- flash attention: one wave per (b,h,16-query tile) ----------------
__global__ __launch_bounds__(32) void attn_kernel(
        const unsigned short* __restrict__ q, const unsigned short* __restrict__ k,
        const unsigned short* __restrict__ vt, const float* __restrict__ tab,
        unsigned short* __restrict__ out) {
    __shared__ __align__(16) unsigned short Pb[16][32];
    int lane = threadIdx.x;
    int hlf = lane >> 4, l16 = lane & 15;
    int qt = blockIdx.x % QTILES;
    int h  = (blockIdx.x / QTILES) % NHEAD;
    int b  = blockIdx.x / (QTILES * NHEAD);
    size_t bh = (size_t)(b * NHEAD + h);

    const unsigned short* qp = q + (bh * NPAD + qt * 16 + l16) * 64;
    Frag qa[2];
#pragma unroll
    for (int t = 0; t < 2; t++) {
        int c0 = t * 32 + hlf * 8;
        qa[t].u[0] = *(const uint4*)(qp + c0);
        qa[t].u[1] = *(const uint4*)(qp + c0 + 16);
    }

    float mrun[8], lrun[8];
    v8f O[4] = {};
#pragma unroll
    for (int r = 0; r < 8; r++) { mrun[r] = -1e30f; lrun[r] = 0.f; }

    for (int c = 0; c < NPAD / 32; c++) {
        int mb = c * 32;
        v8f S[2] = {};
#pragma unroll
        for (int cb = 0; cb < 2; cb++) {
            int mcol = mb + cb * 16 + l16;
            const unsigned short* kp = k + (bh * NPAD + mcol) * 64;
#pragma unroll
            for (int t = 0; t < 2; t++) {
                Frag kb;
                int c0 = t * 32 + hlf * 8;
                kb.u[0] = *(const uint4*)(kp + c0);
                kb.u[1] = *(const uint4*)(kp + c0 + 16);
                S[cb] = __builtin_amdgcn_wmma_f32_16x16x32_bf16(
                    false, qa[t].v, false, kb.v, (short)0, S[cb], false, false);
            }
        }
        // bias + key mask
        float sv[2][8];
#pragma unroll
        for (int cb = 0; cb < 2; cb++) {
            int mkey = mb + cb * 16 + l16;
#pragma unroll
            for (int r = 0; r < 8; r++) {
                int qrow = qt * 16 + r + hlf * 8;
                float s = S[cb][r];
                if (mkey >= NTOK) s = -1e30f;
                else s += relbias(tab, h, (qrow < NTOK) ? qrow : 0, mkey);
                sv[cb][r] = s;
            }
        }
        // online softmax (row reductions across the 16-lane half groups)
#pragma unroll
        for (int r = 0; r < 8; r++) {
            float mx = fmaxf(sv[0][r], sv[1][r]);
            for (int d = 1; d < 16; d <<= 1) mx = fmaxf(mx, __shfl_xor(mx, d, 32));
            float mnew = fmaxf(mrun[r], mx);
            float scl = expf(mrun[r] - mnew);
            mrun[r] = mnew;
            float p0 = expf(sv[0][r] - mnew);
            float p1 = expf(sv[1][r] - mnew);
            float rs = p0 + p1;
            for (int d = 1; d < 16; d <<= 1) rs += __shfl_xor(rs, d, 32);
            lrun[r] = lrun[r] * scl + rs;
#pragma unroll
            for (int fb = 0; fb < 4; fb++) O[fb][r] = O[fb][r] * scl;
            Pb[r + hlf * 8][l16]      = f2bf(p0);   // C-layout -> LDS
            Pb[r + hlf * 8][16 + l16] = f2bf(p1);
        }
        __syncthreads();
        Frag pa;                                    // LDS -> A-layout
        {
            int c0 = hlf * 8;
            pa.u[0] = *(const uint4*)(&Pb[l16][c0]);
            pa.u[1] = *(const uint4*)(&Pb[l16][c0 + 16]);
        }
#pragma unroll
        for (int fb = 0; fb < 4; fb++) {
            Frag vb;
            int f = fb * 16 + l16;
            const unsigned short* vp = vt + (bh * 64 + f) * NPAD + mb + hlf * 8;
            vb.u[0] = *(const uint4*)vp;
            vb.u[1] = *(const uint4*)(vp + 16);
            O[fb] = __builtin_amdgcn_wmma_f32_16x16x32_bf16(
                false, pa.v, false, vb.v, (short)0, O[fb], false, false);
        }
        __syncthreads();
    }

#pragma unroll
    for (int r = 0; r < 8; r++) {
        int qrow = qt * 16 + r + hlf * 8;
        if (qrow >= NTOK) continue;
        float inv = 1.0f / lrun[r];
        size_t base = (size_t)(b * NTOK + qrow) * FDIM + h * 64;
#pragma unroll
        for (int fb = 0; fb < 4; fb++)
            out[base + fb * 16 + l16] = f2bf(O[fb][r] * inv);
    }
}

// ---------------- host launch ----------------
extern "C" void kernel_launch(void* const* d_in, const int* in_sizes, int n_in,
                              void* d_out, int out_size, void* d_ws, size_t ws_size,
                              hipStream_t stream) {
    const float* tokens     = (const float*)d_in[0];
    const float* ln1_g      = (const float*)d_in[1];
    const float* ln1_b      = (const float*)d_in[2];
    const float* q_bias     = (const float*)d_in[3];
    const float* v_bias     = (const float*)d_in[4];
    const float* qkv_w      = (const float*)d_in[5];
    const float* relpos     = (const float*)d_in[6];
    const float* proj_w     = (const float*)d_in[7];
    const float* proj_b     = (const float*)d_in[8];
    const float* scale_attn = (const float*)d_in[9];
    const float* ln2_g      = (const float*)d_in[10];
    const float* ln2_b      = (const float*)d_in[11];
    const float* mlp_w1     = (const float*)d_in[12];
    const float* mlp_b1     = (const float*)d_in[13];
    const float* mlp_w2     = (const float*)d_in[14];
    const float* mlp_b2     = (const float*)d_in[15];
    const float* scale_mlp  = (const float*)d_in[16];
    float* x = (float*)d_out;   // residual stream lives in d_out

    char* ws = (char*)d_ws;
    size_t off = 0;
    auto alloc = [&](size_t bytes) -> void* {
        void* p = ws + off;
        off += (bytes + 255) & ~(size_t)255;
        return p;
    };
    unsigned short* actA = (unsigned short*)alloc((size_t)MTOT * FDIM * 2);
    unsigned short* actH = (unsigned short*)alloc((size_t)MTOT * HIDDEN * 2);
    unsigned short* attn_out = (unsigned short*)alloc((size_t)MTOT * FDIM * 2);
    unsigned short* wbf = (unsigned short*)alloc((size_t)HIDDEN * FDIM * 2);
    size_t qkbytes = (size_t)BATCH * NHEAD * NPAD * 64 * 2;
    unsigned short* qbuf  = (unsigned short*)alloc(qkbytes);
    unsigned short* kbuf  = (unsigned short*)alloc(qkbytes);
    unsigned short* vtbuf = (unsigned short*)alloc(qkbytes);

    hipMemcpyAsync(x, tokens, (size_t)MTOT * FDIM * 4, hipMemcpyDeviceToDevice, stream);
    hipMemsetAsync(qbuf, 0, qkbytes, stream);   // zero padded rows for masking safety
    hipMemsetAsync(kbuf, 0, qkbytes, stream);
    hipMemsetAsync(vtbuf, 0, qkbytes, stream);

    auto cvt = [&](const float* wsrc, size_t n) {
        cvt_kernel<<<(int)(n / 1024), 256, 0, stream>>>(
            (const float4*)wsrc, (uint2*)wbf, (int)(n / 4));
    };

    int mblocks = (MTOT + 127) / 128;   // 37
    for (int l = 0; l < LAYERS; l++) {
        ln_kernel<<<MTOT, 256, 0, stream>>>(x, ln1_g + l * FDIM, ln1_b + l * FDIM, actA);
        cvt(qkv_w + (size_t)l * 3 * FDIM * FDIM, (size_t)3 * FDIM * FDIM);
        gemm_kernel<<<dim3(mblocks, 3 * FDIM / 64), 256, 0, stream>>>(
            actA, wbf, FDIM, 3 * FDIM, 0,
            nullptr, nullptr, nullptr, nullptr,
            q_bias + (size_t)l * NHEAD * FH, v_bias + (size_t)l * NHEAD * FH,
            qbuf, kbuf, vtbuf);
        attn_kernel<<<BATCH * NHEAD * QTILES, 32, 0, stream>>>(
            qbuf, kbuf, vtbuf, relpos + (size_t)l * NRDIST * NHEAD, attn_out);
        cvt(proj_w + (size_t)l * FDIM * FDIM, (size_t)FDIM * FDIM);
        gemm_kernel<<<dim3(mblocks, FDIM / 64), 256, 0, stream>>>(
            attn_out, wbf, FDIM, FDIM, 1,
            proj_b + l * FDIM, scale_attn + l * FDIM, x, nullptr,
            nullptr, nullptr, nullptr, nullptr, nullptr);
        ln_kernel<<<MTOT, 256, 0, stream>>>(x, ln2_g + l * FDIM, ln2_b + l * FDIM, actA);
        cvt(mlp_w1 + (size_t)l * HIDDEN * FDIM, (size_t)HIDDEN * FDIM);
        gemm_kernel<<<dim3(mblocks, HIDDEN / 64), 256, 0, stream>>>(
            actA, wbf, FDIM, HIDDEN, 2,
            mlp_b1 + l * HIDDEN, nullptr, nullptr, actH,
            nullptr, nullptr, nullptr, nullptr, nullptr);
        cvt(mlp_w2 + (size_t)l * FDIM * HIDDEN, (size_t)FDIM * HIDDEN);
        gemm_kernel<<<dim3(mblocks, FDIM / 64), 256, 0, stream>>>(
            actH, wbf, HIDDEN, FDIM, 1,
            mlp_b2 + l * FDIM, scale_mlp + l * FDIM, x, nullptr,
            nullptr, nullptr, nullptr, nullptr, nullptr);
    }
}
